// Gaviko_69784628625772
// MI455X (gfx1250) — matrix operand order repaired and checked by reading, not compiled, not run on hardware
//
#include <hip/hip_runtime.h>
#include <math.h>

// ---------------- model constants ----------------
#define DIMD   768
#define HEADSN 12
#define DHN    64
#define INNERN 768
#define MLPN   3072
#define PTOK   8
#define PLN    20
#define NTOK   1000
#define NGTOK  1009           // P + 1 + N
#define BATCHN 16
#define QKVWN  (3*INNERN)     // 2304
#define LAYERS 2

#define SCALE_MHA  0.125f
#define SCALE_LSA  0.03608439182435161f
#define SCALE_FUSE 0.22360679774997896f

// ---------------- WMMA types ----------------
typedef __bf16 bf16_t;
typedef bf16_t bf16x16 __attribute__((ext_vector_type(16)));
typedef float  f32x8   __attribute__((ext_vector_type(8)));

union FragAB { bf16x16 v; unsigned u[8]; };
union FragC  { f32x8 v;   float    f[8]; };

__device__ __forceinline__ unsigned f2bf(float f) {
  union { float f; unsigned u; } c; c.f = f;
  unsigned u = c.u;
  return (u + 0x7FFFu + ((u >> 16) & 1u)) >> 16;   // RNE to bf16
}
__device__ __forceinline__ unsigned pack2(float a, float b) {
  return (f2bf(a) & 0xFFFFu) | (f2bf(b) << 16);
}
// A-matrix (16x32 bf16) pair index for VGPR j (CDNA5 ISA):
// lanes 0-15: K=0..7 (v0-3), 16..23 (v4-7); lanes 16-31: K=8..15, 24..31
__device__ __forceinline__ int apair(int lane, int j) {
  int kh = (lane >> 4) * 4;
  return (j < 4) ? (kh + j) : (8 + kh + (j - 4));
}
// B-matrix (32x16 bf16): same K order as A: lanes 0-15 K=0..7,16..23; lanes 16-31 K=8..15,24..31
__device__ __forceinline__ int bpair(int lane, int j) {
  int kh = (lane >> 4) * 4;
  return (j < 4) ? (kh + j) : (8 + kh + (j - 4));
}

__device__ __forceinline__ float gelu_exact(float x) {
  return 0.5f * x * (1.0f + erff(x * 0.7071067811865476f));
}
__device__ __forceinline__ float qgelu_f(float x) {
  return x / (1.0f + __expf(-1.702f * x));
}
__device__ __forceinline__ float sigmoid_f(float x) {
  return 1.0f / (1.0f + __expf(-x));
}
// async global->LDS b128 copy (CDNA5 GV mode, tracked by ASYNCcnt)
__device__ __forceinline__ void async_b128(unsigned ldsOff, const void* gaddr) {
  asm volatile("global_load_async_to_lds_b128 %0, %1, off"
               :: "v"(ldsOff), "v"(gaddr) : "memory");
}
__device__ __forceinline__ void wait_async0() {
  asm volatile("s_wait_asynccnt 0" ::: "memory");
}

// =====================================================================
// bf16 conversion kernels (one-shot, keep conversion out of GEMM loops)
// =====================================================================
__global__ __launch_bounds__(256)
void k_cvt_bf16(const float* __restrict__ in, unsigned* __restrict__ out, long npairs)
{
  long i = (long)blockIdx.x * 256 + threadIdx.x;
  if (i >= npairs) return;
  float2 v = ((const float2*)in)[i];
  out[i] = pack2(v.x, v.y);
}

// WT[n*K + k] = bf16(W[k*N + n])  (pre-transpose weights for B-fragment reads)
__global__ __launch_bounds__(256)
void k_cvt_transpose_bf16(const float* __restrict__ W, unsigned short* __restrict__ WT,
                          int K, int N)
{
  long idx = (long)blockIdx.x * 256 + threadIdx.x;
  if (idx >= (long)K * N) return;
  long k = idx / N, n = idx % N;
  WT[n * (long)K + k] = (unsigned short)f2bf(W[idx]);
}

// =====================================================================
// bf16-WMMA GEMM: out = act(A[M,K](bf16) @ WT[N,K](bf16)^T + bias) + res1 + res2
// act: 0=none 1=gelu(exact) 2=qgelu.  outf (f32) and/or outbf (bf16).
// 256 threads = 8 waves (4xM, 2xN), tile 128x64, K-step 32.
// Fast path: global_load_async_to_lds_b128; edge path: guarded scalar.
// =====================================================================
#define BM 128
#define BN 64
#define BK 32

__global__ __launch_bounds__(256)
void k_gemm_wmma(const unsigned short* __restrict__ A,
                 const unsigned short* __restrict__ WT,
                 const float* __restrict__ bias,
                 const float* __restrict__ res1, const float* __restrict__ res2,
                 float* __restrict__ outf, unsigned short* __restrict__ outbf,
                 int M, int N, int K, int act)
{
  __shared__ unsigned lA[BM][16];   // 128 rows x 32 bf16 (64B rows)
  __shared__ unsigned lB[BN][16];   // 64 rows  x 32 bf16

  const int tid  = threadIdx.x;
  const int wave = tid >> 5;
  const int lane = tid & 31;
  const int lrow = lane & 15;
  const int wm   = (wave & 3) * 32;
  const int wn   = (wave >> 2) * 32;
  const int bm   = blockIdx.x * BM;
  const int bn   = blockIdx.y * BN;

  const unsigned ldsA = (unsigned)(unsigned long long)&lA[0][0];
  const unsigned ldsB = (unsigned)(unsigned long long)&lB[0][0];
  const bool krow16 = ((K & 7) == 0);          // rows are 16B-aligned strides

  f32x8 acc[2][2] = {};

  for (int k0 = 0; k0 < K; k0 += BK) {
    const bool fullK = krow16 && (k0 + BK <= K);
    const bool fastA = fullK && (bm + BM <= M);
    const bool fastB = fullK && (bn + BN <= N);

    if (fastA) {
      // 128 rows x 64B = 512 x b128; 2 per thread, DMA'd straight into LDS
      const unsigned short* abase = A + (long)bm * K + k0;
      for (int i = 0; i < 2; ++i) {
        int s = tid * 2 + i;
        int r = s >> 2, q = s & 3;
        async_b128(ldsA + (unsigned)(r * 64 + q * 16),
                   (const void*)(abase + (long)r * K + q * 8));
      }
      if (k0 + 2 * BK <= K)
        __builtin_prefetch(abase + (long)(tid >> 1) * K + BK + (tid & 1) * 16, 0, 1);
    } else {
      for (int i = 0; i < 8; ++i) {
        int u = tid * 8 + i;            // 0..2047
        int r = u >> 4, cp = u & 15;
        int gr = bm + r, gc = k0 + cp * 2;
        unsigned lo = (gr < M && gc     < K) ? (unsigned)A[(long)gr * K + gc]     : 0u;
        unsigned hi = (gr < M && gc + 1 < K) ? (unsigned)A[(long)gr * K + gc + 1] : 0u;
        lA[r][cp] = lo | (hi << 16);
      }
    }
    if (fastB) {
      // 64 rows x 64B = 256 x b128; 1 per thread
      const unsigned short* bbase = WT + (long)bn * K + k0;
      int r = tid >> 2, q = tid & 3;
      async_b128(ldsB + (unsigned)(r * 64 + q * 16),
                 (const void*)(bbase + (long)r * K + q * 8));
    } else {
      for (int i = 0; i < 4; ++i) {
        int u = tid * 4 + i;            // 0..1023
        int n = u >> 4, cp = u & 15;
        int gn = bn + n, gk = k0 + cp * 2;
        unsigned lo = (gn < N && gk     < K) ? (unsigned)WT[(long)gn * K + gk]     : 0u;
        unsigned hi = (gn < N && gk + 1 < K) ? (unsigned)WT[(long)gn * K + gk + 1] : 0u;
        lB[n][cp] = lo | (hi << 16);
      }
    }
    wait_async0();
    __syncthreads();

    FragAB fa[2], fb[2];
    for (int t = 0; t < 2; ++t) {
      int r = wm + t * 16 + lrow;
      for (int j = 0; j < 8; ++j) fa[t].u[j] = lA[r][apair(lane, j)];
    }
    for (int t = 0; t < 2; ++t) {
      int n = wn + t * 16 + lrow;
      for (int j = 0; j < 8; ++j) fb[t].u[j] = lB[n][bpair(lane, j)];
    }
    for (int tm = 0; tm < 2; ++tm)
      for (int tn = 0; tn < 2; ++tn)
        acc[tm][tn] = __builtin_amdgcn_wmma_f32_16x16x32_bf16(
            false, fa[tm].v, false, fb[tn].v, (short)0, acc[tm][tn], false, false);
    __syncthreads();
  }

  for (int tm = 0; tm < 2; ++tm)
    for (int tn = 0; tn < 2; ++tn) {
      int col   = bn + wn + tn * 16 + (lane & 15);
      int rbase = bm + wm + tm * 16 + ((lane >> 4) << 3);
      if (col >= N) continue;
      FragC c; c.v = acc[tm][tn];
      for (int r = 0; r < 8; ++r) {
        int row = rbase + r;
        if (row >= M) continue;
        float v = c.f[r];
        if (bias) v += bias[col];
        if (act == 1) v = gelu_exact(v);
        else if (act == 2) v = qgelu_f(v);
        long idx = (long)row * N + col;
        if (res1) v += res1[idx];
        if (res2) v += res2[idx];
        if (outf)  outf[idx]  = v;
        if (outbf) outbf[idx] = (unsigned short)f2bf(v);
      }
    }
}

// =====================================================================
// LayerNorm: one block per row; optional f32 and/or bf16 outputs
// =====================================================================
__global__ __launch_bounds__(256)
void k_layernorm(const float* __restrict__ in, const float* __restrict__ g,
                 const float* __restrict__ b, float* __restrict__ outf,
                 unsigned short* __restrict__ outbf, int dim)
{
  const int tid = threadIdx.x;
  const long base = (long)blockIdx.x * dim;
  __shared__ float s1[256], s2[256];
  float ls = 0.f, lq = 0.f;
  for (int i = tid; i < dim; i += 256) { float v = in[base + i]; ls += v; lq += v * v; }
  s1[tid] = ls; s2[tid] = lq; __syncthreads();
  for (int o = 128; o > 0; o >>= 1) {
    if (tid < o) { s1[tid] += s1[tid + o]; s2[tid] += s2[tid + o]; }
    __syncthreads();
  }
  float mean = s1[0] / dim;
  float var  = s2[0] / dim - mean * mean;
  float inv  = rsqrtf(var + 1e-5f);
  for (int i = tid; i < dim; i += 256) {
    float v = (in[base + i] - mean) * inv * g[i] + b[i];
    if (outf)  outf[base + i]  = v;
    if (outbf) outbf[base + i] = (unsigned short)f2bf(v);
  }
}

// =====================================================================
// MHA flash attention (bf16 WMMA). 1 wave per (b, h, 16-query tile).
// qkv: bf16 [B, NG, 3*INNER]; out o2: bf16 [B, NG, INNER]
// =====================================================================
__global__ __launch_bounds__(32)
void k_mha_flash(const unsigned short* __restrict__ qkv, unsigned short* __restrict__ o2)
{
  const int qt   = blockIdx.x * 16;
  const int h    = blockIdx.y;
  const int b    = blockIdx.z;
  const int lane = threadIdx.x;
  const int lrow = lane & 15;

  __shared__ unsigned sQ[16][32];   // 16 q-rows x 64 bf16 (dh pairs)
  __shared__ unsigned sK[32][32];   // 32 keys   x 64 bf16
  __shared__ unsigned sV[64][16];   // 64 dh     x 32 keys (key pairs)
  __shared__ float    sS[16][33];
  __shared__ unsigned sP[16][17];
  __shared__ float    sM[16], sL[16], sScale[16];

  const unsigned short* qbase = qkv + (long)b * NGTOK * QKVWN;

  // ---- stage Q tile (b128 copies) ----
  {
    int r = lane >> 1, seg = lane & 1;
    int qrow = qt + r;
    if (qt + 16 <= NGTOK) {
      const uint4* src = (const uint4*)(qbase + (long)qrow * QKVWN + h * DHN) + seg * 4;
      uint4* dst = (uint4*)&sQ[r][seg * 16];
      dst[0] = src[0]; dst[1] = src[1]; dst[2] = src[2]; dst[3] = src[3];
    } else {
      const unsigned* src = (qrow < NGTOK)
          ? (const unsigned*)(qbase + (long)qrow * QKVWN + h * DHN) : (const unsigned*)0;
      for (int i = 0; i < 16; ++i) sQ[r][seg * 16 + i] = src ? src[seg * 16 + i] : 0u;
    }
  }
  if (lane < 16) { sM[lane] = -1e30f; sL[lane] = 0.f; }
  __syncthreads();

  FragAB qf[2];
  for (int s = 0; s < 2; ++s)
    for (int j = 0; j < 8; ++j)
      qf[s].u[j] = sQ[lrow][s * 16 + apair(lane, j)];

  f32x8 accO[4] = {};
  const int nkt = (NGTOK + 31) / 32;
  for (int kt = 0; kt < nkt; ++kt) {
    const int kbase = kt * 32;
    const bool full = (kbase + 32 <= NGTOK);

    // ---- stage K tile: 1 key row per lane (b128 copies) ----
    {
      int key = kbase + lane;
      if (full) {
        const uint4* src = (const uint4*)(qbase + (long)key * QKVWN + INNERN + h * DHN);
        uint4* dst = (uint4*)&sK[lane][0];
        for (int i = 0; i < 8; ++i) dst[i] = src[i];
      } else {
        const unsigned* src = (key < NGTOK)
            ? (const unsigned*)(qbase + (long)key * QKVWN + INNERN + h * DHN) : (const unsigned*)0;
        for (int i = 0; i < 32; ++i) sK[lane][i] = src ? src[i] : 0u;
      }
    }
    // ---- stage V tile transposed to key-pairs ----
    {
      int kp = lane & 15, ch = lane >> 4;   // ch selects dh half (32 elems)
      int key0 = kbase + kp * 2, key1 = key0 + 1;
      unsigned r0[16], r1[16];
      if (full) {
        const uint4* s0 = (const uint4*)(qbase + (long)key0 * QKVWN + 2 * INNERN + h * DHN + ch * 32);
        const uint4* s1 = (const uint4*)(qbase + (long)key1 * QKVWN + 2 * INNERN + h * DHN + ch * 32);
        for (int i = 0; i < 4; ++i) {
          uint4 a = s0[i]; uint4 bb = s1[i];
          r0[i*4+0]=a.x; r0[i*4+1]=a.y; r0[i*4+2]=a.z; r0[i*4+3]=a.w;
          r1[i*4+0]=bb.x; r1[i*4+1]=bb.y; r1[i*4+2]=bb.z; r1[i*4+3]=bb.w;
        }
      } else {
        const unsigned* s0 = (key0 < NGTOK)
            ? (const unsigned*)(qbase + (long)key0 * QKVWN + 2 * INNERN + h * DHN + ch * 32) : (const unsigned*)0;
        const unsigned* s1 = (key1 < NGTOK)
            ? (const unsigned*)(qbase + (long)key1 * QKVWN + 2 * INNERN + h * DHN + ch * 32) : (const unsigned*)0;
        for (int i = 0; i < 16; ++i) { r0[i] = s0 ? s0[i] : 0u; r1[i] = s1 ? s1[i] : 0u; }
      }
      for (int i = 0; i < 16; ++i) {
        int c = ch * 32 + 2 * i;
        sV[c][kp]     = (r0[i] & 0xFFFFu) | (r1[i] << 16);
        sV[c + 1][kp] = (r0[i] >> 16)     | (r1[i] & 0xFFFF0000u);
      }
    }
    __syncthreads();

    // ---- S = Q K^T (two 16x16 tiles) ----
    f32x8 sAcc[2] = {};
    for (int st = 0; st < 2; ++st)
      for (int s = 0; s < 2; ++s) {
        FragAB kf;
        for (int j = 0; j < 8; ++j)
          kf.u[j] = sK[st * 16 + lrow][s * 16 + bpair(lane, j)];
        sAcc[st] = __builtin_amdgcn_wmma_f32_16x16x32_bf16(
            false, qf[s].v, false, kf.v, (short)0, sAcc[st], false, false);
      }
    for (int st = 0; st < 2; ++st) {
      FragC c; c.v = sAcc[st];
      int coln = st * 16 + (lane & 15);
      int key  = kbase + coln;
      int rb   = (lane >> 4) << 3;
      for (int r = 0; r < 8; ++r)
        sS[rb + r][coln] = (key < NGTOK) ? c.f[r] * SCALE_MHA : -1e30f;
    }
    __syncthreads();

    // ---- online softmax (one lane per query row) ----
    if (lane < 16) {
      float m_old = sM[lane], l_old = sL[lane];
      float tmax = -1e30f;
      for (int j = 0; j < 32; ++j) tmax = fmaxf(tmax, sS[lane][j]);
      float m_new = fmaxf(m_old, tmax);
      float scale = __expf(m_old - m_new);
      float lsum = 0.f;
      for (int j = 0; j < 16; ++j) {
        float p0 = __expf(sS[lane][2 * j]     - m_new);
        float p1 = __expf(sS[lane][2 * j + 1] - m_new);
        lsum += p0 + p1;
        sP[lane][j] = pack2(p0, p1);
      }
      sM[lane] = m_new; sL[lane] = l_old * scale + lsum; sScale[lane] = scale;
    }
    __syncthreads();

    {
      int rb = (lane >> 4) << 3;
      for (int t = 0; t < 4; ++t) {
        FragC c; c.v = accO[t];
        for (int r = 0; r < 8; ++r) c.f[r] *= sScale[rb + r];
        accO[t] = c.v;
      }
    }
    FragAB pf;
    for (int j = 0; j < 8; ++j) pf.u[j] = sP[lrow][apair(lane, j)];

    for (int t = 0; t < 4; ++t) {
      FragAB vf;
      for (int j = 0; j < 8; ++j)
        vf.u[j] = sV[t * 16 + lrow][bpair(lane, j)];
      accO[t] = __builtin_amdgcn_wmma_f32_16x16x32_bf16(
          false, pf.v, false, vf.v, (short)0, accO[t], false, false);
    }
    __syncthreads();
  }

  for (int t = 0; t < 4; ++t) {
    FragC c; c.v = accO[t];
    int col = t * 16 + (lane & 15);
    int rb  = (lane >> 4) << 3;
    for (int r = 0; r < 8; ++r) {
      int row = qt + rb + r;
      if (row < NGTOK)
        o2[((long)b * NGTOK + row) * INNERN + h * DHN + col] =
            (unsigned short)f2bf(c.f[r] / sL[rb + r]);
    }
  }
}

// =====================================================================
// Local self-attention (masked, 20-dim latent). 1 block per (q,b).
// qkvl: f32 [B,N,60]; outv: bf16 [B,N,20]
// =====================================================================
__global__ __launch_bounds__(256)
void k_lsa_attn(const float* __restrict__ qkvl, const float* __restrict__ mask,
                unsigned short* __restrict__ outv)
{
  const int q = blockIdx.x, b = blockIdx.y, tid = threadIdx.x;
  __shared__ float sq[PLN];
  __shared__ float red[256];
  __shared__ float spo[256 * PLN];
  if (tid < PLN) sq[tid] = qkvl[((long)b * NTOK + q) * 60 + tid];
  __syncthreads();

  float lmax = -1e30f;
  for (int n = tid; n < NTOK; n += 256) {
    const float* kr = &qkvl[((long)b * NTOK + n) * 60 + PLN];
    float s = 0.f;
    for (int d = 0; d < PLN; ++d) s += sq[d] * kr[d];
    s = s * SCALE_LSA + mask[(long)q * NTOK + n];
    lmax = fmaxf(lmax, s);
  }
  red[tid] = lmax; __syncthreads();
  for (int o = 128; o > 0; o >>= 1) { if (tid < o) red[tid] = fmaxf(red[tid], red[tid + o]); __syncthreads(); }
  float gmax = red[0]; __syncthreads();

  float po[PLN]; for (int d = 0; d < PLN; ++d) po[d] = 0.f;
  float lsum = 0.f;
  for (int n = tid; n < NTOK; n += 256) {
    const float* kr = &qkvl[((long)b * NTOK + n) * 60 + PLN];
    const float* vr = &qkvl[((long)b * NTOK + n) * 60 + 2 * PLN];
    float s = 0.f;
    for (int d = 0; d < PLN; ++d) s += sq[d] * kr[d];
    s = s * SCALE_LSA + mask[(long)q * NTOK + n];
    float p = __expf(s - gmax);
    lsum += p;
    for (int d = 0; d < PLN; ++d) po[d] += p * vr[d];
  }
  red[tid] = lsum; __syncthreads();
  for (int o = 128; o > 0; o >>= 1) { if (tid < o) red[tid] += red[tid + o]; __syncthreads(); }
  float gsum = red[0];
  for (int d = 0; d < PLN; ++d) spo[tid * PLN + d] = po[d];
  __syncthreads();
  if (tid < PLN) {
    float a = 0.f;
    for (int t = 0; t < 256; ++t) a += spo[t * PLN + tid];
    outv[((long)b * NTOK + q) * PLN + tid] = (unsigned short)f2bf(a / gsum);
  }
}

// =====================================================================
// CLS gate: per batch.  imp[B,8], gw[B]
// =====================================================================
__global__ __launch_bounds__(64)
void k_cls_gate(const float* __restrict__ xlat,
                const float* __restrict__ cg, const float* __restrict__ cb,
                const float* __restrict__ w1, const float* __restrict__ b1,
                const float* __restrict__ w2, const float* __restrict__ b2,
                const float* __restrict__ gg, const float* __restrict__ gb,
                const float* __restrict__ glw, const float* __restrict__ glb,
                float* __restrict__ imp, float* __restrict__ gw)
{
  const int b = blockIdx.x, tid = threadIdx.x;
  __shared__ float clsn[PLN], clsng[PLN], hh[64];
  const float* c0 = xlat + ((long)b * NGTOK + PTOK) * PLN;
  if (tid == 0) {
    float m = 0.f; for (int d = 0; d < PLN; ++d) m += c0[d]; m /= PLN;
    float v = 0.f; for (int d = 0; d < PLN; ++d) { float x = c0[d] - m; v += x * x; }
    float inv = rsqrtf(v / PLN + 1e-5f);
    for (int d = 0; d < PLN; ++d) {
      float xn = (c0[d] - m) * inv;
      clsn[d]  = xn * cg[d] + cb[d];
      clsng[d] = xn * gg[d] + gb[d];
    }
  }
  __syncthreads();
  {
    float a = b1[tid];
    for (int d = 0; d < PLN; ++d) a += clsn[d] * w1[d * 64 + tid];
    hh[tid] = gelu_exact(a);
  }
  __syncthreads();
  if (tid < PTOK) {
    float a = b2[tid];
    for (int j = 0; j < 64; ++j) a += hh[j] * w2[j * PTOK + tid];
    imp[b * PTOK + tid] = sigmoid_f(a);
  }
  if (tid == 0) {
    float a = glb[0];
    for (int d = 0; d < PLN; ++d) a += clsng[d] * glw[d];
    gw[b] = sigmoid_f(a);
  }
}

// =====================================================================
// Prompt fusion cross-attention: 1 block per (p, b).  outc: f32 [B,8,20]
// =====================================================================
__global__ __launch_bounds__(256)
void k_fuse(const float* __restrict__ xlat, const float* __restrict__ tok,
            long tokBatchStride, int tokRowOff,
            const float* __restrict__ qw, const float* __restrict__ qb,
            float* __restrict__ outc)
{
  const int p = blockIdx.x, b = blockIdx.y, tid = threadIdx.x;
  __shared__ float qq[PLN];
  __shared__ float red[256];
  __shared__ float spo[256 * PLN];
  const float* pr = xlat + ((long)b * NGTOK + p) * PLN;
  if (tid < PLN) {
    float a = qb[tid];
    for (int e = 0; e < PLN; ++e) a += pr[e] * qw[e * PLN + tid];
    qq[tid] = a;
  }
  __syncthreads();
  const float* tb = tok + (long)b * tokBatchStride + (long)tokRowOff * PLN;

  float lmax = -1e30f;
  for (int n = tid; n < NTOK; n += 256) {
    const float* t = tb + (long)n * PLN;
    float s = 0.f;
    for (int d = 0; d < PLN; ++d) s += qq[d] * t[d];
    lmax = fmaxf(lmax, s * SCALE_FUSE);
  }
  red[tid] = lmax; __syncthreads();
  for (int o = 128; o > 0; o >>= 1) { if (tid < o) red[tid] = fmaxf(red[tid], red[tid + o]); __syncthreads(); }
  float gmax = red[0]; __syncthreads();

  float po[PLN]; for (int d = 0; d < PLN; ++d) po[d] = 0.f;
  float lsum = 0.f;
  for (int n = tid; n < NTOK; n += 256) {
    const float* t = tb + (long)n * PLN;
    float s = 0.f;
    for (int d = 0; d < PLN; ++d) s += qq[d] * t[d];
    float pw = __expf(s * SCALE_FUSE - gmax);
    lsum += pw;
    for (int d = 0; d < PLN; ++d) po[d] += pw * t[d];
  }
  red[tid] = lsum; __syncthreads();
  for (int o = 128; o > 0; o >>= 1) { if (tid < o) red[tid] += red[tid + o]; __syncthreads(); }
  float gsum = red[0];
  for (int d = 0; d < PLN; ++d) spo[tid * PLN + d] = po[d];
  __syncthreads();
  if (tid < PLN) {
    float a = 0.f;
    for (int t = 0; t < 256; ++t) a += spo[t * PLN + tid];
    outc[((long)b * PTOK + p) * PLN + tid] = a / gsum;
  }
}

// =====================================================================
// comb = [enh (8 rows), cls, gimg] -> bf16 [B, NG, 20]
// =====================================================================
__global__ __launch_bounds__(256)
void k_comb(const float* __restrict__ xlat, const float* __restrict__ gc,
            const float* __restrict__ lc, const float* __restrict__ imp,
            const float* __restrict__ gwv, unsigned short* __restrict__ comb)
{
  const int b = blockIdx.y;
  const int idx = blockIdx.x * 256 + threadIdx.x;
  if (idx >= NGTOK * PLN) return;
  const int row = idx / PLN, d = idx % PLN;
  float v;
  if (row < PTOK) {
    float g_ = gwv[b];
    float f  = g_ * gc[((long)b * PTOK + row) * PLN + d]
             + (1.f - g_) * lc[((long)b * PTOK + row) * PLN + d];
    v = f * imp[b * PTOK + row];
  } else {
    v = xlat[((long)b * NGTOK + row) * PLN + d];
  }
  comb[((long)b * NGTOK + row) * PLN + d] = (unsigned short)f2bf(v);
}

// =====================================================================
// host side
// =====================================================================
static inline void launch_gemm(hipStream_t s, const unsigned short* A,
                               const unsigned short* WT, const float* bias,
                               const float* r1, const float* r2,
                               float* outf, unsigned short* outbf,
                               int M, int N, int K, int act)
{
  dim3 grid((M + BM - 1) / BM, (N + BN - 1) / BN);
  k_gemm_wmma<<<grid, 256, 0, s>>>(A, WT, bias, r1, r2, outf, outbf, M, N, K, act);
}
static inline void launch_cvtT(hipStream_t s, const float* W, unsigned short* WT, int K, int N)
{
  long total = (long)K * N;
  k_cvt_transpose_bf16<<<(unsigned)((total + 255) / 256), 256, 0, s>>>(W, WT, K, N);
}
static inline void launch_cvt(hipStream_t s, const float* in, unsigned short* out, long n)
{
  long np = n / 2;
  k_cvt_bf16<<<(unsigned)((np + 255) / 256), 256, 0, s>>>(in, (unsigned*)out, np);
}

extern "C" void kernel_launch(void* const* d_in, const int* in_sizes, int n_in,
                              void* d_out, int out_size, void* d_ws, size_t ws_size,
                              hipStream_t stream)
{
  (void)in_sizes; (void)n_in; (void)out_size; (void)ws_size;
  const float* in_g = (const float*)d_in[0];
  const float* in_l = (const float*)d_in[1];
  int pi = 2;
  const float* p_lsa_norm_g = (const float*)d_in[pi++];
  const float* p_lsa_norm_b = (const float*)d_in[pi++];
  const float* p_lsa_down_w = (const float*)d_in[pi++];
  const float* p_lsa_down_b = (const float*)d_in[pi++];
  const float* p_lsa_qkv_w  = (const float*)d_in[pi++];
  const float* p_lsa_up_w   = (const float*)d_in[pi++];
  const float* p_lsa_up_b   = (const float*)d_in[pi++];
  const float* p_attn_norm_g= (const float*)d_in[pi++];
  const float* p_attn_norm_b= (const float*)d_in[pi++];
  const float* p_attn_qkv_w = (const float*)d_in[pi++];
  const float* p_attn_out_w = (const float*)d_in[pi++];
  const float* p_attn_out_b = (const float*)d_in[pi++];
  const float* p_mlp_norm_g = (const float*)d_in[pi++];
  const float* p_mlp_norm_b = (const float*)d_in[pi++];
  const float* p_mlp_w1     = (const float*)d_in[pi++];
  const float* p_mlp_b1     = (const float*)d_in[pi++];
  const float* p_mlp_w2     = (const float*)d_in[pi++];
  const float* p_mlp_b2     = (const float*)d_in[pi++];
  const float* p_ap_down_w  = (const float*)d_in[pi++];
  const float* p_ap_down_b  = (const float*)d_in[pi++];
  const float* p_ap_up_w    = (const float*)d_in[pi++];
  const float* p_ap_up_b    = (const float*)d_in[pi++];
  const float* p_cls_ln_g   = (const float*)d_in[pi++];
  const float* p_cls_ln_b   = (const float*)d_in[pi++];
  const float* p_cls_w1     = (const float*)d_in[pi++];
  const float* p_cls_b1     = (const float*)d_in[pi++];
  const float* p_cls_w2     = (const float*)d_in[pi++];
  const float* p_cls_b2     = (const float*)d_in[pi++];
  const float* p_gl_ln_g    = (const float*)d_in[pi++];
  const float* p_gl_ln_b    = (const float*)d_in[pi++];
  const float* p_gl_w       = (const float*)d_in[pi++];
  const float* p_gl_b       = (const float*)d_in[pi++];
  const float* p_gq_w       = (const float*)d_in[pi++];
  const float* p_gq_b       = (const float*)d_in[pi++];
  const float* p_lq_w       = (const float*)d_in[pi++];
  const float* p_lq_b       = (const float*)d_in[pi++];
  const float* p_final_g    = (const float*)d_in[pi++];
  const float* p_final_b    = (const float*)d_in[pi++];
  const float* mask         = (const float*)d_in[pi++];

  const int MG = BATCHN * NGTOK;   // 16144
  const int ML = BATCHN * NTOK;    // 16000

  char* wsb = (char*)d_ws;
  size_t off = 0;
  auto alloc = [&](size_t bytes) {
    size_t a = (off + 255) & ~(size_t)255;
    off = a + bytes;
    return (void*)(wsb + a);
  };
  // f32 buffers
  float* g      = (float*)alloc(sizeof(float) * (size_t)MG * DIMD);
  float* lf     = (float*)alloc(sizeof(float) * (size_t)ML * DIMD);
  float* xlat   = (float*)alloc(sizeof(float) * (size_t)MG * PLN);
  float* loclat = (float*)alloc(sizeof(float) * (size_t)ML * PLN);
  float* qkvl   = (float*)alloc(sizeof(float) * (size_t)ML * 3 * PLN);
  float* prompt = (float*)alloc(sizeof(float) * (size_t)MG * DIMD);
  float* impb   = (float*)alloc(sizeof(float) * BATCHN * PTOK);
  float* gwb    = (float*)alloc(sizeof(float) * BATCHN);
  float* gcb    = (float*)alloc(sizeof(float) * BATCHN * PTOK * PLN);
  float* lcb    = (float*)alloc(sizeof(float) * BATCHN * PTOK * PLN);
  // bf16 buffers
  unsigned short* xnbf   = (unsigned short*)alloc(2ull * MG * DIMD);
  unsigned short* qkvbf  = (unsigned short*)alloc(2ull * MG * QKVWN);
  unsigned short* o2bf   = (unsigned short*)alloc(2ull * MG * INNERN);
  unsigned short* hbf    = (unsigned short*)alloc(2ull * MG * MLPN);
  unsigned short* gbf    = (unsigned short*)alloc(2ull * MG * DIMD);
  unsigned short* lfbf   = (unsigned short*)alloc(2ull * ML * DIMD);
  unsigned short* xlbf   = (unsigned short*)alloc(2ull * ML * PLN);
  unsigned short* attnvbf= (unsigned short*)alloc(2ull * ML * PLN);
  unsigned short* combbf = (unsigned short*)alloc(2ull * MG * PLN);
  unsigned short* wt     = (unsigned short*)alloc(2ull * DIMD * MLPN);  // max K*N

  hipMemcpyAsync(g,  in_g, sizeof(float) * (size_t)MG * DIMD, hipMemcpyDeviceToDevice, stream);
  hipMemcpyAsync(lf, in_l, sizeof(float) * (size_t)ML * DIMD, hipMemcpyDeviceToDevice, stream);

  for (int i = 0; i < LAYERS; ++i) {
    const float* lsa_ng = p_lsa_norm_g + (size_t)i * DIMD;
    const float* lsa_nb = p_lsa_norm_b + (size_t)i * DIMD;
    const float* lsa_dw = p_lsa_down_w + (size_t)i * DIMD * PLN;
    const float* lsa_db = p_lsa_down_b + (size_t)i * PLN;
    const float* lsa_qw = p_lsa_qkv_w  + (size_t)i * PLN * 3 * PLN;
    const float* lsa_uw = p_lsa_up_w   + (size_t)i * PLN * DIMD;
    const float* lsa_ub = p_lsa_up_b   + (size_t)i * DIMD;
    const float* att_ng = p_attn_norm_g+ (size_t)i * DIMD;
    const float* att_nb = p_attn_norm_b+ (size_t)i * DIMD;
    const float* att_qw = p_attn_qkv_w + (size_t)i * DIMD * QKVWN;
    const float* att_ow = p_attn_out_w + (size_t)i * INNERN * DIMD;
    const float* att_ob = p_attn_out_b + (size_t)i * DIMD;
    const float* mlp_ng = p_mlp_norm_g + (size_t)i * DIMD;
    const float* mlp_nb = p_mlp_norm_b + (size_t)i * DIMD;
    const float* mlp_w1 = p_mlp_w1     + (size_t)i * DIMD * MLPN;
    const float* mlp_b1 = p_mlp_b1     + (size_t)i * MLPN;
    const float* mlp_w2 = p_mlp_w2     + (size_t)i * MLPN * DIMD;
    const float* mlp_b2 = p_mlp_b2     + (size_t)i * DIMD;
    const float* ap_dw  = p_ap_down_w  + (size_t)i * DIMD * PLN;
    const float* ap_db  = p_ap_down_b  + (size_t)i * PLN;
    const float* ap_uw  = p_ap_up_w    + (size_t)i * PLN * DIMD;
    const float* ap_ub  = p_ap_up_b    + (size_t)i * DIMD;
    const float* cls_g  = p_cls_ln_g   + (size_t)i * PLN;
    const float* cls_b  = p_cls_ln_b   + (size_t)i * PLN;
    const float* cls_w1 = p_cls_w1     + (size_t)i * PLN * 64;
    const float* cls_b1 = p_cls_b1     + (size_t)i * 64;
    const float* cls_w2 = p_cls_w2     + (size_t)i * 64 * PTOK;
    const float* cls_b2 = p_cls_b2     + (size_t)i * PTOK;
    const float* gl_g   = p_gl_ln_g    + (size_t)i * PLN;
    const float* gl_b   = p_gl_ln_b    + (size_t)i * PLN;
    const float* gl_w   = p_gl_w       + (size_t)i * PLN;
    const float* gl_bb  = p_gl_b       + (size_t)i * 1;
    const float* gq_w   = p_gq_w       + (size_t)i * PLN * PLN;
    const float* gq_b   = p_gq_b       + (size_t)i * PLN;
    const float* lq_w   = p_lq_w       + (size_t)i * PLN * PLN;
    const float* lq_b   = p_lq_b       + (size_t)i * PLN;

    // ---- LSA block ----
    k_layernorm<<<ML, 256, 0, stream>>>(lf, lsa_ng, lsa_nb, (float*)0, xnbf, DIMD);
    launch_cvtT(stream, lsa_dw, wt, DIMD, PLN);
    launch_gemm(stream, xnbf, wt, lsa_db, 0, 0, (float*)0, xlbf, ML, PLN, DIMD, 0);
    launch_cvtT(stream, lsa_qw, wt, PLN, 3 * PLN);
    launch_gemm(stream, xlbf, wt, 0, 0, 0, qkvl, (unsigned short*)0, ML, 3 * PLN, PLN, 0);
    { dim3 grid(NTOK, BATCHN);
      k_lsa_attn<<<grid, 256, 0, stream>>>(qkvl, mask, attnvbf); }
    launch_cvtT(stream, lsa_uw, wt, PLN, DIMD);
    launch_gemm(stream, attnvbf, wt, lsa_ub, lf, 0, lf, (unsigned short*)0, ML, DIMD, PLN, 0);

    // ---- MHA block ----
    k_layernorm<<<MG, 256, 0, stream>>>(g, att_ng, att_nb, (float*)0, xnbf, DIMD);
    launch_cvtT(stream, att_qw, wt, DIMD, QKVWN);
    launch_gemm(stream, xnbf, wt, 0, 0, 0, (float*)0, qkvbf, MG, QKVWN, DIMD, 0);
    { dim3 grid((NGTOK + 15) / 16, HEADSN, BATCHN);
      k_mha_flash<<<grid, 32, 0, stream>>>(qkvbf, o2bf); }
    launch_cvtT(stream, att_ow, wt, INNERN, DIMD);
    launch_gemm(stream, o2bf, wt, att_ob, g, 0, g, (unsigned short*)0, MG, DIMD, INNERN, 0);

    // ---- adapter / prompt path ----
    launch_cvt(stream, g,  gbf,  (long)MG * DIMD);
    launch_cvt(stream, lf, lfbf, (long)ML * DIMD);
    launch_cvtT(stream, ap_dw, wt, DIMD, PLN);
    launch_gemm(stream, gbf,  wt, ap_db, 0, 0, xlat,   (unsigned short*)0, MG, PLN, DIMD, 2);
    launch_gemm(stream, lfbf, wt, ap_db, 0, 0, loclat, (unsigned short*)0, ML, PLN, DIMD, 2);
    k_cls_gate<<<BATCHN, 64, 0, stream>>>(xlat, cls_g, cls_b, cls_w1, cls_b1,
                                          cls_w2, cls_b2, gl_g, gl_b, gl_w, gl_bb,
                                          impb, gwb);
    { dim3 grid(PTOK, BATCHN);
      k_fuse<<<grid, 256, 0, stream>>>(xlat, xlat, (long)NGTOK * PLN, PTOK + 1,
                                       gq_w, gq_b, gcb);
      k_fuse<<<grid, 256, 0, stream>>>(xlat, loclat, (long)NTOK * PLN, 0,
                                       lq_w, lq_b, lcb); }
    { dim3 grid((NGTOK * PLN + 255) / 256, BATCHN);
      k_comb<<<grid, 256, 0, stream>>>(xlat, gcb, lcb, impb, gwb, combbf); }
    launch_cvtT(stream, ap_uw, wt, PLN, DIMD);
    launch_gemm(stream, combbf, wt, ap_ub, 0, 0, prompt, (unsigned short*)0, MG, DIMD, PLN, 0);

    // ---- MLP block ----
    k_layernorm<<<MG, 256, 0, stream>>>(g, mlp_ng, mlp_nb, (float*)0, xnbf, DIMD);
    launch_cvtT(stream, mlp_w1, wt, DIMD, MLPN);
    launch_gemm(stream, xnbf, wt, mlp_b1, 0, 0, (float*)0, hbf, MG, MLPN, DIMD, 1);
    launch_cvtT(stream, mlp_w2, wt, MLPN, DIMD);
    launch_gemm(stream, hbf, wt, mlp_b2, g, prompt, g, (unsigned short*)0, MG, DIMD, MLPN, 0);
  }

  // ---- final LN into f32 output ----
  k_layernorm<<<MG, 256, 0, stream>>>(g, p_final_g, p_final_b, (float*)d_out,
                                      (unsigned short*)0, DIMD);
}
